// RWKV6Attention_10093173145935
// MI455X (gfx1250) — compile-verified, matrix operand order checked by reading, not run on hardware
//
#include <hip/hip_runtime.h>
#include <hip/hip_bf16.h>

// ---------------- types / helpers ----------------
typedef __attribute__((ext_vector_type(16))) __bf16 v16bf;
typedef __attribute__((ext_vector_type(8)))  float  v8f;
typedef __attribute__((ext_vector_type(4)))  unsigned int v4u;

union V16 { v16bf v; v4u u2[2]; };

__device__ __forceinline__ unsigned short f2bf(float f) {
  unsigned u = __builtin_bit_cast(unsigned, f);
  u += 0x7FFFu + ((u >> 16) & 1u);          // round-to-nearest-even
  return (unsigned short)(u >> 16);
}
__device__ __forceinline__ float bf2f(unsigned short h) {
  unsigned u = ((unsigned)h) << 16;
  return __builtin_bit_cast(float, u);
}

// Load one 16x32 WMMA operand fragment from a [rows][contraction]-major bf16
// array. Per CDNA5 ISA layout: lanes 0-15 hold row m with K {0..7,16..23},
// lanes 16-31 hold row m with K {8..15,24..31}; each half is 8 contiguous
// bf16 (16B) along the contraction dim -> two b128 loads per lane.
__device__ __forceinline__ v16bf ld_frag(const unsigned short* base, int row0,
                                         int ld, int k0) {
  int lane = threadIdx.x & 31;
  int half = lane >> 4;
  const unsigned short* p = base + (size_t)(row0 + (lane & 15)) * ld + k0 + half * 8;
  V16 x;
  x.u2[0] = *(const v4u*)p;
  x.u2[1] = *(const v4u*)(p + 16);
  return x.v;
}

__device__ __forceinline__ v8f wmma_bf16(v16bf a, v16bf b, v8f c) {
  return __builtin_amdgcn_wmma_f32_16x16x32_bf16(false, a, false, b, (short)0,
                                                 c, false, false);
}

// ---------------- small elementwise kernels ----------------
__global__ void cvt_bf16_kernel(const float* __restrict__ in,
                                unsigned short* __restrict__ out, int n) {
  int i = blockIdx.x * blockDim.x + threadIdx.x;
  if (i < n) out[i] = f2bf(in[i]);
}

// xz = bf16(x + (shift(x) - x) * mu_x)
__global__ void token_mix0_kernel(const float* __restrict__ hs,
                                  const float* __restrict__ mu_x,
                                  unsigned short* __restrict__ xz,
                                  int BT, int T, int D) {
  int i = blockIdx.x * blockDim.x + threadIdx.x;
  if (i >= BT * D) return;
  int d = i % D;
  int tok = i / D;
  float x = hs[i];
  float prev = (tok % T) ? hs[i - D] : 0.f;
  xz[i] = f2bf(x + (prev - x) * mu_x[d]);
}

// ---------------- generic WMMA GEMM: C[M,N] = A[M,K] @ B[N,K]^T -------------
// Each wave owns a 16x(16*NT) output strip; the A fragment is reused across
// NT WMMAs. NT and the epilogue are compile-time so the K-loop has zero
// control flow (no exec predication, no accumulator arrays -> no spills).
// EPI: 0 -> f32 store, 1 -> bf16 store, 2 -> tanh -> bf16, 3 -> +bias[col] f32
template <int NT, int EPI>
__device__ __forceinline__ void gemm_store_tile(void* C, int ldc, int tm,
                                                int tn, const v8f& c,
                                                const float* bias) {
  int lane = threadIdx.x & 31;
  int half = lane >> 4;
  int col = tn * 16 + (lane & 15);
  if (EPI == 0 || EPI == 3) {
    float* out = (float*)C;
    float bv = (EPI == 3) ? bias[col] : 0.f;
#pragma unroll
    for (int i = 0; i < 8; ++i) {
      int row = tm * 16 + (half ? i + 8 : i);
      out[(size_t)row * ldc + col] = c[i] + bv;
    }
  } else {
    unsigned short* out = (unsigned short*)C;
#pragma unroll
    for (int i = 0; i < 8; ++i) {
      int row = tm * 16 + (half ? i + 8 : i);
      float v = c[i];
      if (EPI == 2) v = tanhf(v);
      out[(size_t)row * ldc + col] = f2bf(v);
    }
  }
}

template <int NT, int EPI>
__global__ void gemm_wmma_kernel(const unsigned short* __restrict__ A, int lda,
                                 const unsigned short* __restrict__ B, int ldb,
                                 void* __restrict__ C, int ldc,
                                 int M, int N, int K,
                                 const float* __restrict__ bias) {
  int wid = threadIdx.x >> 5;
  int tilesN = N >> 4;
  int strips = tilesN / NT;               // exact by construction
  int strip = blockIdx.x * 8 + wid;
  if (strip >= (M >> 4) * strips) return;
  int tm = strip / strips, sn = strip % strips;
  int tn0 = sn * NT;

  v8f c0 = {}, c1 = {}, c2 = {}, c3 = {};
#pragma unroll 2
  for (int k0 = 0; k0 < K; k0 += 32) {
    v16bf a = ld_frag(A, tm * 16, lda, k0);
    c0 = wmma_bf16(a, ld_frag(B, (tn0 + 0) * 16, ldb, k0), c0);
    if (NT > 1) c1 = wmma_bf16(a, ld_frag(B, (tn0 + 1) * 16, ldb, k0), c1);
    if (NT > 2) c2 = wmma_bf16(a, ld_frag(B, (tn0 + 2) * 16, ldb, k0), c2);
    if (NT > 3) c3 = wmma_bf16(a, ld_frag(B, (tn0 + 3) * 16, ldb, k0), c3);
  }
  gemm_store_tile<NT, EPI>(C, ldc, tm, tn0 + 0, c0, bias);
  if (NT > 1) gemm_store_tile<NT, EPI>(C, ldc, tm, tn0 + 1, c1, bias);
  if (NT > 2) gemm_store_tile<NT, EPI>(C, ldc, tm, tn0 + 2, c2, bias);
  if (NT > 3) gemm_store_tile<NT, EPI>(C, ldc, tm, tn0 + 3, c3, bias);
}

// ------------- mus + lerp mix: mixed_i = x + delta*(z@Wx2_i^T + bias_i) -----
__global__ void mus_mix_kernel(const unsigned short* __restrict__ z,   // [BT,160]
                               const unsigned short* __restrict__ Wx2, // [1024,160]
                               const float* __restrict__ xbias,        // [5,1024]
                               const float* __restrict__ hs,           // [BT,1024]
                               unsigned short* __restrict__ mixed,     // 5x[BT,1024]
                               int BT, int T) {
  const int D = 1024, LR = 32;
  int wid = threadIdx.x >> 5, lane = threadIdx.x & 31;
  int tile = blockIdx.x * 8 + wid;
  int tilesN = D >> 4;
  if (tile >= (BT >> 4) * tilesN) return;
  int tm = tile / tilesN, tn = tile % tilesN;
  int col = tn * 16 + (lane & 15);
  int half = lane >> 4;
  float xv[8], dv[8];
#pragma unroll
  for (int i = 0; i < 8; ++i) {
    int row = tm * 16 + (half ? i + 8 : i);
    float x = hs[(size_t)row * D + col];
    float prev = (row % T) ? hs[(size_t)(row - 1) * D + col] : 0.f;
    xv[i] = x; dv[i] = prev - x;
  }
  for (int p = 0; p < 5; ++p) {
    v8f c = {};
    c = wmma_bf16(ld_frag(z, tm * 16, 5 * LR, p * LR),
                  ld_frag(Wx2, tn * 16, 5 * LR, p * LR), c);
    float bia = xbias[p * D + col];
    unsigned short* out = mixed + (size_t)p * BT * D;
#pragma unroll
    for (int i = 0; i < 8; ++i) {
      int row = tm * 16 + (half ? i + 8 : i);
      float mu = c[i] + bia;
      out[(size_t)row * D + col] = f2bf(xv[i] + dv[i] * mu);
    }
  }
}

// ------------- chunked RWKV6 recurrence (one workgroup per (b,h)) -----------
// Chunk C=32; within a chunk everything is bf16 WMMA:
//   A      = tril(r~ k~^T) + diag((r*u).k)   [32,32]
//   o      = A @ v + r~ @ S0^T               [32,256]
//   S^T    = diag(e^{sum w}) S0^T + v^T k^   [256,128]  (state kept transposed)
#define CH 32
__global__ __launch_bounds__(256, 1)
void rwkv_chunk_kernel(const unsigned short* __restrict__ rbuf, // [BT,512] bf16
                       const unsigned short* __restrict__ kbuf, // [BT,512] bf16
                       const float* __restrict__ wbuf,          // [BT,512] f32 (LoRA out, pre -exp)
                       const unsigned short* __restrict__ vbuf, // [BT,1024] bf16
                       const float* __restrict__ bonus,         // [4,128]
                       float* __restrict__ obuf,                // [BT,1024] f32
                       int T) {
  const int K = 128, V = 256, HK = 512, HV = 1024;
  __shared__ __align__(16) unsigned short ST[V * K];     // state^T [v][k], 64KB
  __shared__ __align__(16) unsigned short rt[CH * K];    // r~ [t][k]
  __shared__ __align__(16) unsigned short kt[CH * K];    // k~ [t][k]
  __shared__ __align__(16) unsigned short khT[K * CH];   // k^ [k][t]
  __shared__ __align__(16) unsigned short vT[V * CH];    // v^T [v][t]
  __shared__ __align__(16) unsigned short Amat[CH * CH]; // masked attn [t][s]
  __shared__ float logp[CH * K];                         // staged -exp(w), then cumsum
  __shared__ float pC[K];
  __shared__ float pCe[K];
  __shared__ float diag[CH];

  int tid = threadIdx.x, wid = tid >> 5, lane = tid & 31, half = lane >> 4;
  int bh = blockIdx.x, b = bh >> 2, h = bh & 3;
  size_t base = (size_t)b * T;

  for (int i = tid; i < V * K; i += 256) ST[i] = 0;
  __syncthreads();

  for (int c = 0; c < T / CH; ++c) {
    int t0 = c * CH;
    // phase 1a: stage per-step log-decay coalesced into LDS (all threads)
    for (int e = tid; e < CH * K; e += 256) {
      int t = e >> 7, kk = e & 127;
      logp[e] = -__expf(wbuf[(base + t0 + t) * HK + h * K + kk]);
    }
    __syncthreads();
    // phase 1b: in-place inclusive cumsum along t, entirely in LDS
    if (tid < K) {
      int kk = tid;
      float acc = 0.f;
      for (int t = 0; t < CH; ++t) {
        acc += logp[t * K + kk];
        logp[t * K + kk] = acc;
      }
      pC[kk] = acc;
      pCe[kk] = __expf(acc);
    }
    __syncthreads();
    // phase 2: build decay-scaled bf16 operands in LDS
    for (int e = tid; e < CH * K; e += 256) {
      int t = e >> 7, kk = e & 127;
      size_t g = (base + t0 + t) * HK + h * K + kk;
      float rv = bf2f(rbuf[g]);
      float kv = bf2f(kbuf[g]);
      float lp = logp[t * K + kk];
      float lpp = t ? logp[(t - 1) * K + kk] : 0.f;
      rt[e] = f2bf(rv * __expf(lpp));
      kt[e] = f2bf(kv * __expf(-lp));
      khT[kk * CH + t] = f2bf(kv * __expf(pC[kk] - lp));
    }
    for (int e = tid; e < CH * V; e += 256) {
      int t = e >> 8, vc = e & 255;
      vT[vc * CH + t] = vbuf[(base + t0 + t) * HV + h * V + vc];
    }
    if (tid < CH) {
      int t = tid;
      size_t g = (base + t0 + t) * HK + h * K;
      float s = 0.f;
      for (int kk = 0; kk < K; ++kk)
        s += bf2f(rbuf[g + kk]) * bonus[h * K + kk] * bf2f(kbuf[g + kk]);
      diag[t] = s;
    }
    if (c + 1 < T / CH && tid < CH)  // prefetch next chunk's decay rows
      __builtin_prefetch(&wbuf[(base + t0 + CH + tid) * HK + h * K], 0, 1);
    __syncthreads();
    // phase 3: A = tril(r~ k~^T) strictly-lower + bonus diagonal
    if (wid < 4) {
      int ti = wid >> 1, si = wid & 1;
      v8f a = {};
#pragma unroll
      for (int ks = 0; ks < 4; ++ks)
        a = wmma_bf16(ld_frag(rt, ti * 16, K, ks * 32),
                      ld_frag(kt, si * 16, K, ks * 32), a);
      int ss = si * 16 + (lane & 15);
#pragma unroll
      for (int i = 0; i < 8; ++i) {
        int tt = ti * 16 + (half ? i + 8 : i);
        float v = (ss < tt) ? a[i] : (ss == tt ? diag[tt] : 0.f);
        Amat[tt * CH + ss] = f2bf(v);
      }
    }
    __syncthreads();
    // phase 4: o = A @ v + r~ @ S0^T  (32 output tiles, 4 per wave)
    for (int tile = wid; tile < 2 * (V / 16); tile += 8) {
      int ti = tile >> 4, vi = tile & 15;
      v8f o = {};
      o = wmma_bf16(ld_frag(Amat, ti * 16, CH, 0),
                    ld_frag(vT, vi * 16, CH, 0), o);
#pragma unroll
      for (int ks = 0; ks < 4; ++ks)
        o = wmma_bf16(ld_frag(rt, ti * 16, K, ks * 32),
                      ld_frag(ST, vi * 16, K, ks * 32), o);
      int vc = vi * 16 + (lane & 15);
#pragma unroll
      for (int i = 0; i < 8; ++i) {
        int tt = ti * 16 + (half ? i + 8 : i);
        obuf[(base + t0 + tt) * HV + h * V + vc] = o[i];
      }
    }
    __syncthreads();
    // phase 5: S^T = diag(pCe) * S0^T + v^T @ k^  (128 tiles, 16 per wave)
    for (int tile = wid; tile < (V / 16) * (K / 16); tile += 8) {
      int vi = tile >> 3, ki = tile & 7;
      int kk = ki * 16 + (lane & 15);
      v8f s;
#pragma unroll
      for (int i = 0; i < 8; ++i) {
        int vr = vi * 16 + (half ? i + 8 : i);
        s[i] = pCe[kk] * bf2f(ST[vr * K + kk]);
      }
      s = wmma_bf16(ld_frag(vT, vi * 16, CH, 0),
                    ld_frag(khT, ki * 16, CH, 0), s);
#pragma unroll
      for (int i = 0; i < 8; ++i) {
        int vr = vi * 16 + (half ? i + 8 : i);
        ST[vr * K + kk] = f2bf(s[i]);
      }
    }
    __syncthreads();
  }
}

// ------------- per-head GroupNorm + SiLU gate (wave32 reductions) -----------
__global__ void gn_gate_kernel(const float* __restrict__ obuf,
                               const unsigned short* __restrict__ gbuf,
                               const float* __restrict__ gnw,
                               const float* __restrict__ gnb,
                               unsigned short* __restrict__ gated, int BT) {
  const int V = 256, HV = 1024;
  int wid = threadIdx.x >> 5, lane = threadIdx.x & 31;
  int gid = blockIdx.x * 8 + wid;
  if (gid >= BT * 4) return;
  int tok = gid >> 2, h = gid & 3;
  const float* op = obuf + (size_t)tok * HV + h * V;
  float vals[8], s = 0.f, sq = 0.f;
#pragma unroll
  for (int j = 0; j < 8; ++j) {
    float x = op[lane + j * 32];
    vals[j] = x; s += x; sq += x * x;
  }
#pragma unroll
  for (int off = 16; off; off >>= 1) {
    s += __shfl_xor(s, off, 32);
    sq += __shfl_xor(sq, off, 32);
  }
  float mean = s * (1.f / V);
  float var = sq * (1.f / V) - mean * mean;
  float rstd = rsqrtf(var + 1e-5f);
#pragma unroll
  for (int j = 0; j < 8; ++j) {
    int idx = h * V + lane + j * 32;
    float y = (vals[j] - mean) * rstd * gnw[idx] + gnb[idx];
    float g = bf2f(gbuf[(size_t)tok * HV + idx]);
    y *= g / (1.f + __expf(-g));
    gated[(size_t)tok * HV + idx] = f2bf(y);
  }
}

// ------------------------------ launcher ------------------------------------
static void launch_gemm(const unsigned short* A, int lda,
                        const unsigned short* B, int ldb,
                        void* C, int ldc, int M, int N, int K, int epi,
                        const float* bias, hipStream_t stream) {
  int tilesN = N >> 4;
  int nt = (tilesN % 4 == 0) ? 4 : 2;
  int strips = (M >> 4) * (tilesN / nt);
  int blocks = (strips + 7) / 8;
#define GEMM_CASE(NTV, EPIV)                                                  \
  gemm_wmma_kernel<NTV, EPIV><<<blocks, 256, 0, stream>>>(A, lda, B, ldb, C,  \
                                                          ldc, M, N, K, bias)
  if (nt == 4) {
    switch (epi) {
      case 0: GEMM_CASE(4, 0); break;
      case 1: GEMM_CASE(4, 1); break;
      case 2: GEMM_CASE(4, 2); break;
      default: GEMM_CASE(4, 3); break;
    }
  } else {
    switch (epi) {
      case 0: GEMM_CASE(2, 0); break;
      case 1: GEMM_CASE(2, 1); break;
      case 2: GEMM_CASE(2, 2); break;
      default: GEMM_CASE(2, 3); break;
    }
  }
#undef GEMM_CASE
}

extern "C" void kernel_launch(void* const* d_in, const int* in_sizes, int n_in,
                              void* d_out, int out_size, void* d_ws, size_t ws_size,
                              hipStream_t stream) {
  const float* hs    = (const float*)d_in[0];
  const float* mu_x  = (const float*)d_in[1];
  const float* W_x1  = (const float*)d_in[2];
  const float* W_x2  = (const float*)d_in[3];
  const float* xbias = (const float*)d_in[4];
  const float* W_r   = (const float*)d_in[5];
  const float* W_k   = (const float*)d_in[6];
  const float* W_v   = (const float*)d_in[7];
  const float* W_g   = (const float*)d_in[8];
  const float* A_w   = (const float*)d_in[9];
  const float* B_w   = (const float*)d_in[10];
  const float* b_w   = (const float*)d_in[11];
  const float* bonus = (const float*)d_in[12];
  const float* gnw   = (const float*)d_in[13];
  const float* gnb   = (const float*)d_in[14];
  const float* W_o   = (const float*)d_in[15];

  const int B = 8, T = 2048, D = 1024, BT = B * T;
  const int KD = 512, VD = 1024, LR5 = 160, GLR = 64;

  char* ws = (char*)d_ws;
  size_t off = 0;
  auto alloc = [&](size_t bytes) -> char* {
    char* p = ws + off;
    off += (bytes + 255) & ~(size_t)255;
    return p;
  };
  unsigned short* wb_x1 = (unsigned short*)alloc((size_t)LR5 * D * 2);
  unsigned short* wb_x2 = (unsigned short*)alloc((size_t)D * LR5 * 2);
  unsigned short* wb_r  = (unsigned short*)alloc((size_t)KD * D * 2);
  unsigned short* wb_k  = (unsigned short*)alloc((size_t)KD * D * 2);
  unsigned short* wb_v  = (unsigned short*)alloc((size_t)VD * D * 2);
  unsigned short* wb_g  = (unsigned short*)alloc((size_t)VD * D * 2);
  unsigned short* wb_aw = (unsigned short*)alloc((size_t)GLR * D * 2);
  unsigned short* wb_bw = (unsigned short*)alloc((size_t)KD * GLR * 2);
  unsigned short* wb_o  = (unsigned short*)alloc((size_t)D * VD * 2);
  unsigned short* xz    = (unsigned short*)alloc((size_t)BT * D * 2);
  unsigned short* zbuf  = (unsigned short*)alloc((size_t)BT * LR5 * 2);
  unsigned short* mixed = (unsigned short*)alloc((size_t)5 * BT * D * 2);
  unsigned short* rbuf  = (unsigned short*)alloc((size_t)BT * KD * 2);
  unsigned short* kbuf  = (unsigned short*)alloc((size_t)BT * KD * 2);
  unsigned short* wtmp  = (unsigned short*)alloc((size_t)BT * GLR * 2);
  float*          wdec  = (float*)alloc((size_t)BT * KD * 4);
  unsigned short* vbuf  = (unsigned short*)alloc((size_t)BT * VD * 2);
  unsigned short* gbuf  = (unsigned short*)alloc((size_t)BT * VD * 2);
  float*          obuf  = (float*)alloc((size_t)BT * VD * 4);
  unsigned short* gated = (unsigned short*)alloc((size_t)BT * VD * 2);

  auto cvt = [&](const float* src, unsigned short* dst, int n) {
    cvt_bf16_kernel<<<(n + 255) / 256, 256, 0, stream>>>(src, dst, n);
  };
  cvt(W_x1, wb_x1, LR5 * D);
  cvt(W_x2, wb_x2, D * LR5);
  cvt(W_r, wb_r, KD * D);
  cvt(W_k, wb_k, KD * D);
  cvt(W_v, wb_v, VD * D);
  cvt(W_g, wb_g, VD * D);
  cvt(A_w, wb_aw, GLR * D);
  cvt(B_w, wb_bw, KD * GLR);
  cvt(W_o, wb_o, D * VD);

  token_mix0_kernel<<<(BT * D + 255) / 256, 256, 0, stream>>>(hs, mu_x, xz, BT, T, D);

  // z = tanh(xz @ W_x1^T)   (N=160 -> NT=2 path)
  launch_gemm(xz, D, wb_x1, D, zbuf, LR5, BT, LR5, D, 2, nullptr, stream);
  // mixed_{r,w,k,v,g} = x + delta * (z @ W_x2_i^T + x_bias_i)
  {
    int tiles = (BT >> 4) * (D >> 4);
    mus_mix_kernel<<<(tiles + 7) / 8, 256, 0, stream>>>(zbuf, wb_x2, xbias, hs,
                                                        mixed, BT, T);
  }
  unsigned short* mr = mixed + (size_t)0 * BT * D;
  unsigned short* mw = mixed + (size_t)1 * BT * D;
  unsigned short* mk = mixed + (size_t)2 * BT * D;
  unsigned short* mv = mixed + (size_t)3 * BT * D;
  unsigned short* mg = mixed + (size_t)4 * BT * D;

  launch_gemm(mr, D, wb_r, D, rbuf, KD, BT, KD, D, 1, nullptr, stream);      // r
  launch_gemm(mw, D, wb_aw, D, wtmp, GLR, BT, GLR, D, 2, nullptr, stream);   // tanh LoRA
  launch_gemm(wtmp, GLR, wb_bw, GLR, wdec, KD, BT, KD, GLR, 3, b_w, stream); // w (f32)
  launch_gemm(mk, D, wb_k, D, kbuf, KD, BT, KD, D, 1, nullptr, stream);      // k
  launch_gemm(mv, D, wb_v, D, vbuf, VD, BT, VD, D, 1, nullptr, stream);      // v
  launch_gemm(mg, D, wb_g, D, gbuf, VD, BT, VD, D, 1, nullptr, stream);      // g

  // chunked linear-attention recurrence: one workgroup per (b, head)
  rwkv_chunk_kernel<<<32, 256, 0, stream>>>(rbuf, kbuf, wdec, vbuf, bonus,
                                            obuf, T);

  // GroupNorm per head + SiLU gate
  gn_gate_kernel<<<(BT * 4 + 7) / 8, 256, 0, stream>>>(obuf, gbuf, gnw, gnb,
                                                       gated, BT);

  // out = gated @ W_o^T  (f32 to d_out)
  launch_gemm(gated, VD, wb_o, VD, (float*)d_out, D, BT, D, VD, 0, nullptr,
              stream);
}
